// LabelGCN_10823317586755
// MI455X (gfx1250) — matrix operand order repaired and checked by reading, not compiled, not run on hardware
//
#include <hip/hip_runtime.h>
#include <hip/hip_bf16.h>

#define BROWS 16384
#define DDIM  1024
#define CDIM  1000
#define CPAD  1024

typedef __attribute__((ext_vector_type(16))) __bf16 bf16x16;
typedef __attribute__((ext_vector_type(8)))  float  f32x8;
typedef __attribute__((ext_vector_type(4)))  unsigned u32x4;
typedef __attribute__((ext_vector_type(4)))  int      i32x4;
typedef __attribute__((ext_vector_type(8)))  int      i32x8;

struct Q2 { uint4 lo, hi; };

__device__ __forceinline__ bf16x16 make_frag(uint4 lo, uint4 hi) {
    Q2 q; q.lo = lo; q.hi = hi;
    return __builtin_bit_cast(bf16x16, q);
}

// f32 -> bf16 round-to-nearest-even (bit-level)
__device__ __forceinline__ unsigned short f2bf(float f) {
    unsigned u = __builtin_bit_cast(unsigned, f);
    u += 0x7fffu + ((u >> 16) & 1u);
    return (unsigned short)(u >> 16);
}

// TDM: DMA a 2D bf16 tile [rows=16, rowLen=1024 elts] from global (row stride 1024 elts)
// into LDS at byte offset lds_off. D# built per CDNA5 ISA §8.3/8.4.
__device__ __forceinline__ void tdm_load_b_tile(const unsigned short* gsrc, unsigned lds_off) {
    unsigned long long ga = (unsigned long long)(uintptr_t)gsrc;
    u32x4 g0;
    g0[0] = 1u;                                              // count=1, user descriptor
    g0[1] = lds_off;                                         // lds_addr [63:32]
    g0[2] = (unsigned)(ga & 0xffffffffu);                    // global_addr[31:0]
    g0[3] = (unsigned)((ga >> 32) & 0x01ffffffu) | (2u << 30); // global_addr[56:32] | type=2
    i32x8 g1;
    g1[0] = (int)(1u << 16);          // workgroup_mask=0, data_size=1 (2 bytes)
    g1[1] = (int)(1024u << 16);       // tensor_dim0[15:0]=1024 at bits[63:48]
    g1[2] = (int)(1024u << 16);       // tensor_dim0 hi=0 | tensor_dim1[15:0]=1024
    g1[3] = (int)(1024u << 16);       // tensor_dim1 hi=0 | tile_dim0=1024
    g1[4] = 16;                       // tile_dim1=16, tile_dim2=0 (2D)
    g1[5] = 1024;                     // tensor_dim0_stride low32 = 1024 elts
    g1[6] = 0;
    g1[7] = 0;
    i32x4 z = {0, 0, 0, 0};
#if defined(__clang_major__) && (__clang_major__ >= 23)
    i32x8 z8 = {0, 0, 0, 0, 0, 0, 0, 0};
    __builtin_amdgcn_tensor_load_to_lds(g0, g1, z, z, z8, 0);
#else
    __builtin_amdgcn_tensor_load_to_lds(g0, g1, z, z, 0);
#endif
}

// fp32 [rows, cols] row-major -> bf16 [rowsPad, ldDst] row-major, zero padded
__global__ void pad_convert(const float* __restrict__ src, unsigned short* __restrict__ dst,
                            int rows, int cols, int rowsPad, int ldDst) {
    int idx = blockIdx.x * blockDim.x + threadIdx.x;
    int total = rowsPad * ldDst;
    if (idx >= total) return;
    int r = idx / ldDst;
    int c = idx - r * ldDst;
    unsigned short v = 0;
    if (r < rows && c < cols) v = f2bf(src[(size_t)r * cols + c]);
    dst[idx] = v;
}

// NT GEMM: acc[m,n] = sum_k A[m,k] * Bm[n,k], A/Bm bf16 row-major (ld = 1024), fp32 accum.
// B tile (16 x 1024) staged to LDS by the Tensor Data Mover once per block.
// MODE 0: v = acc + bp[n];  dout[m,n] = v;  tout[m,n] = bf16(v * w1[n])
// MODE 1: v = relu(acc);                    tout[m,n] = bf16(v * w2[n])
// MODE 2: dout[m,n] += 0.5f * acc
template<int MODE>
__global__ void __launch_bounds__(256)
gemm_wmma(const unsigned short* __restrict__ A,
          const unsigned short* __restrict__ Bm,
          float* __restrict__ dout,
          unsigned short* __restrict__ tout,
          const float* __restrict__ bias,
          const float* __restrict__ hw) {
    __shared__ __align__(32) unsigned short Bsh[16 * DDIM];   // 32 KB, LDS offset 0

    const int lane = threadIdx.x & 31;   // wave32
    const int wave = threadIdx.x >> 5;   // 8 waves / block
    const int nl   = lane & 15;
    const int half = lane >> 4;

    const int m0 = blockIdx.x * 128 + wave * 16;  // wave's 16 output rows
    const int n0 = blockIdx.y * 16;               // block's 16 output cols

    // Stage B tile (rows n0..n0+15, full K) into LDS with one TDM op from wave 0.
    if (wave == 0) {
        tdm_load_b_tile(Bm + (size_t)n0 * DDIM, 0u);
        __builtin_amdgcn_s_wait_tensorcnt(0);
    }
    __syncthreads();

    // A fragment (16x32 bf16): lane = row m0+nl; half-wave 0 covers K {0..7,16..23},
    // half-wave 1 covers K {8..15,24..31}  -> base offset half*8, second load at +16.
    const unsigned short* Ap = A + (size_t)(m0 + nl) * DDIM + half * 8;
    // B fragment (32x16 bf16) from LDS: lane = column n0+nl (LDS row nl);
// half-wave 0 covers K {0..7,16..23}, half-wave 1 covers K {8..15,24..31}.
    const unsigned short* Bl = &Bsh[nl * DDIM + half * 8];

    f32x8 acc = {0.f, 0.f, 0.f, 0.f, 0.f, 0.f, 0.f, 0.f};

#pragma unroll 4
    for (int k = 0; k < DDIM; k += 32) {
        __builtin_prefetch(Ap + k + 256, 0, 0);   // global_prefetch_b8
        bf16x16 a = make_frag(*(const uint4*)(Ap + k), *(const uint4*)(Ap + k + 16));
        bf16x16 b = make_frag(*(const uint4*)(Bl + k), *(const uint4*)(Bl + k + 16));
        acc = __builtin_amdgcn_wmma_f32_16x16x32_bf16(
            /*neg_a=*/false, a, /*neg_b=*/false, b,
            /*c_mod=*/(short)0, acc, /*reuse_a=*/false, /*reuse_b=*/false);
    }

    const int n = n0 + nl;
    float w = 0.f, bb = 0.f;
    if (MODE == 0 || MODE == 1) {
        if (n < CDIM) w = (hw[n] + hw[CDIM + n] + hw[2 * CDIM + n]) * (1.0f / 3.0f);
    }
    if (MODE == 0) {
        if (n < CDIM) bb = bias[n];
    }

    // C/D layout: VGPR r holds row m0 + r (lanes 0-15) or m0 + r + 8 (lanes 16-31)
#pragma unroll
    for (int r = 0; r < 8; ++r) {
        const int m = m0 + r + half * 8;
        float v = acc[r];
        if (MODE == 0) {
            v += bb;
            if (n < CDIM) dout[(size_t)m * CDIM + n] = v;
            tout[(size_t)m * CPAD + n] = f2bf(v * w);     // w==0 pads cols >= 1000
        } else if (MODE == 1) {
            v = v > 0.f ? v : 0.f;
            tout[(size_t)m * CPAD + n] = f2bf(v * w);
        } else {
            if (n < CDIM) {
                size_t o = (size_t)m * CDIM + n;
                dout[o] = dout[o] + 0.5f * v;
            }
        }
    }
}

extern "C" void kernel_launch(void* const* d_in, const int* in_sizes, int n_in,
                              void* d_out, int out_size, void* d_ws, size_t ws_size,
                              hipStream_t stream) {
    (void)in_sizes; (void)n_in; (void)out_size; (void)ws_size;
    const float* x   = (const float*)d_in[0];
    const float* adj = (const float*)d_in[1];
    const float* Wp  = (const float*)d_in[2];
    const float* bp  = (const float*)d_in[3];
    const float* hw1 = (const float*)d_in[4];
    const float* hw2 = (const float*)d_in[5];
    float* out = (float*)d_out;

    // Workspace layout (bf16, elements): xb/t2 alias (xb dead after gemm<0>)
    unsigned short* xb   = (unsigned short*)d_ws;               // 16384*1024
    unsigned short* t1   = xb  + (size_t)BROWS * DDIM;          // 16384*1024
    unsigned short* Wpb  = t1  + (size_t)BROWS * DDIM;          // 1024*1024
    unsigned short* adjb = Wpb + (size_t)CPAD * DDIM;           // 1024*1024
    unsigned short* t2   = xb;

    {   // x -> bf16 (exact-size, no pad)
        int total = BROWS * DDIM;
        pad_convert<<<(total + 255) / 256, 256, 0, stream>>>(x, xb, BROWS, DDIM, BROWS, DDIM);
    }
    {   // Wp [1000,1024] -> [1024,1024] bf16, zero rows 1000..1023
        int total = CPAD * DDIM;
        pad_convert<<<(total + 255) / 256, 256, 0, stream>>>(Wp, Wpb, CDIM, DDIM, CPAD, DDIM);
    }
    {   // adj [1000,1000] -> [1024,1024] bf16, zero padded
        int total = CPAD * CPAD;
        pad_convert<<<(total + 255) / 256, 256, 0, stream>>>(adj, adjb, CDIM, CDIM, CPAD, CPAD);
    }

    dim3 grid(BROWS / 128, CPAD / 16);   // 128 x 64 blocks, 8 waves each, 16x16 tile/wave
    gemm_wmma<0><<<grid, 256, 0, stream>>>(xb, Wpb,  out, t1,      bp,      hw1);
    gemm_wmma<1><<<grid, 256, 0, stream>>>(t1, adjb, nullptr, t2,  nullptr, hw2);
    gemm_wmma<2><<<grid, 256, 0, stream>>>(t2, adjb, out, nullptr, nullptr, nullptr);
}
